// ASRRNN_25683904430324
// MI455X (gfx1250) — compile-verified, matrix operand order checked by reading, not run on hardware
//
#include <hip/hip_runtime.h>
#include <hip/hip_bf16.h>

// ---------- plain ext-vector types ----------
typedef __attribute__((ext_vector_type(16))) __bf16    v16bf;
typedef __attribute__((ext_vector_type(8)))  float     v8f;
typedef __attribute__((ext_vector_type(4)))  unsigned  u32x4;
typedef __attribute__((ext_vector_type(4)))  int       i32x4;

union Frag { u32x4 u[2]; v16bf v; };

#define LDSK 40   // LDS row pitch in halfs (80B) -> conflict-free b128 frag reads

// ---------- gfx1250 async copy to LDS (guarded; falls back to sync copy) ----
#if defined(__gfx1250__) && __has_builtin(__builtin_amdgcn_global_load_async_to_lds_b128) && __has_builtin(__builtin_amdgcn_s_wait_asynccnt)
#define USE_ASYNC 1
#else
#define USE_ASYNC 0
#endif

__device__ __forceinline__ void cp16(void* lds, const void* g) {
#if USE_ASYNC
  __builtin_amdgcn_global_load_async_to_lds_b128(
      (__attribute__((address_space(1))) i32x4*)(void*)g,
      (__attribute__((address_space(3))) i32x4*)lds, 0, 0);
#else
  *(u32x4*)lds = *(const u32x4*)g;
#endif
}
__device__ __forceinline__ void cp16_wait() {
#if USE_ASYNC
  __builtin_amdgcn_s_wait_asynccnt(0);
#endif
}

// ===========================================================================
// conv over channels + relu, LDS transpose -> x[(t*8+b)][f] in f32 AND bf16
// fused: [B=8][C=16][F=1024][T=512] f32
// ===========================================================================
__global__ __launch_bounds__(256) void conv_relu(const float* __restrict__ fused,
                                                 const float* __restrict__ w,
                                                 const float* __restrict__ cb,
                                                 float* __restrict__ out,
                                                 __bf16* __restrict__ outbf) {
  __shared__ float tile[32][33];
  const int tx = threadIdx.x & 31, ty = threadIdx.x >> 5;
  const int t0 = blockIdx.x * 32, f0 = blockIdx.y * 32, b = blockIdx.z;
  float wreg[16];
#pragma unroll
  for (int c = 0; c < 16; ++c) wreg[c] = w[c];
  const float bias = cb[0];
#pragma unroll
  for (int i = 0; i < 4; ++i) {
    const int fl = ty + i * 8;
    const size_t f = (size_t)f0 + fl;
    float acc = 0.f;
#pragma unroll
    for (int c = 0; c < 16; ++c)
      acc += wreg[c] * fused[(((size_t)b * 16 + c) * 1024 + f) * 512 + t0 + tx];
    acc += bias;
    tile[fl][tx] = acc > 0.f ? acc : 0.f;
  }
  __syncthreads();
#pragma unroll
  for (int i = 0; i < 4; ++i) {
    const int tl = ty + i * 8;
    const unsigned o = ((unsigned)(t0 + tl) * 8u + b) * 1024u + f0 + tx;
    const float v = tile[tx][tl];
    out[o] = v;
    outbf[o] = (__bf16)v;
  }
}

// ===========================================================================
// W[K][N] f32  ->  Wt[N][K] bf16   (tiled transpose + convert)
// ===========================================================================
__global__ __launch_bounds__(256) void transpose_convert(const float* __restrict__ in,
                                                         __bf16* __restrict__ out,
                                                         int K, int N) {
  __shared__ float tile[32][33];
  const int tx = threadIdx.x & 31, ty = threadIdx.x >> 5;
  const int n0 = blockIdx.x * 32, k0 = blockIdx.y * 32;
#pragma unroll
  for (int i = 0; i < 4; ++i) {
    const int kl = ty + i * 8;
    tile[kl][tx] = in[(size_t)(k0 + kl) * N + n0 + tx];
  }
  __syncthreads();
#pragma unroll
  for (int i = 0; i < 4; ++i) {
    const int nl = ty + i * 8;
    out[(size_t)(n0 + nl) * K + k0 + tx] = (__bf16)tile[tx][nl];
  }
}

// ===========================================================================
// C[M][N](f32) = A[M][K](bf16) * Bt[N][K](bf16)^T
// WG: 256 thr = 8 waves, tile 128x128; wave = 32x64 = 2x4 WMMA 16x16 tiles.
// Ping-pong LDS, async global->LDS copies overlapped with WMMA.
// REMAP: C[((row&7)*(M/8) + (row>>3))*N + col] for the final FC (b,t,k) layout.
// ===========================================================================
template <bool HAS_BIAS, bool REMAP>
__global__ __launch_bounds__(256) void gemm_bf16(const __bf16* __restrict__ A,
                                                 const __bf16* __restrict__ Bt,
                                                 float* __restrict__ C,
                                                 const float* __restrict__ bias,
                                                 int M, int N, int K) {
  __shared__ __bf16 As[2][128 * LDSK];
  __shared__ __bf16 Bs[2][128 * LDSK];
  const int tid  = threadIdx.x;
  const int lane = tid & 31, wid = tid >> 5;
  const int wm = wid & 3, wn = wid >> 2;               // 4x2 wave grid
  const int rowBase = blockIdx.x * 128;
  const int colBase = blockIdx.y * 128;

  v8f zero = {0.f, 0.f, 0.f, 0.f, 0.f, 0.f, 0.f, 0.f};
  v8f acc[2][4];
#pragma unroll
  for (int mt = 0; mt < 2; ++mt)
#pragma unroll
    for (int nt = 0; nt < 4; ++nt) acc[mt][nt] = zero;

  const int krow = (lane >> 4) << 3;  // 0 or 8 (ISA 16-bit A-frag K split)
  const int lr   = lane & 15;
  const int nk   = K >> 5;

  // tile loader: 128x32 bf16 = 512 x 16B chunks each for A and Bt
  auto load_tiles = [&](int kt, int buf) {
    const int k0 = kt << 5;
#pragma unroll
    for (int j = 0; j < 2; ++j) {
      const int idx = tid + j * 256;
      const int r = idx >> 2, c8 = (idx & 3) << 3;
      cp16(&As[buf][r * LDSK + c8], A  + (unsigned)(rowBase + r) * K + k0 + c8);
      cp16(&Bs[buf][r * LDSK + c8], Bt + (unsigned)(colBase + r) * K + k0 + c8);
    }
  };

  load_tiles(0, 0);
  cp16_wait();
  __syncthreads();

  for (int kt = 0; kt < nk; ++kt) {
    const int cur = kt & 1;
    if (kt + 1 < nk) load_tiles(kt + 1, cur ^ 1);

    v16bf afrag[2], bfrag[4];
#pragma unroll
    for (int mt = 0; mt < 2; ++mt) {
      const __bf16* p = &As[cur][(wm * 32 + mt * 16 + lr) * LDSK + krow];
      Frag f; f.u[0] = *(const u32x4*)p; f.u[1] = *(const u32x4*)(p + 16);
      afrag[mt] = f.v;
    }
#pragma unroll
    for (int nt = 0; nt < 4; ++nt) {
      const __bf16* p = &Bs[cur][(wn * 64 + nt * 16 + lr) * LDSK + krow];
      Frag f; f.u[0] = *(const u32x4*)p; f.u[1] = *(const u32x4*)(p + 16);
      bfrag[nt] = f.v;
    }
#pragma unroll
    for (int mt = 0; mt < 2; ++mt)
#pragma unroll
      for (int nt = 0; nt < 4; ++nt)
        acc[mt][nt] = __builtin_amdgcn_wmma_f32_16x16x32_bf16(
            false, afrag[mt], false, bfrag[nt], (short)0, acc[mt][nt],
            false, false);

    if (kt + 1 < nk) cp16_wait();
    __syncthreads();
  }

  // ---- epilogue: C layout (VGPR i -> M = i + 8*(lane>=16), N = lane&15) ----
  const int mhi  = (lane >> 4) << 3;
  const int nl   = lane & 15;
  const int Tdim = M >> 3;
#pragma unroll
  for (int nt = 0; nt < 4; ++nt) {
    const int col = colBase + wn * 64 + nt * 16 + nl;
    float bv = 0.f;
    if (HAS_BIAS) bv = bias[col];   // compile-time dead when !HAS_BIAS
#pragma unroll
    for (int mt = 0; mt < 2; ++mt) {
      const int rb = rowBase + wm * 32 + mt * 16 + mhi;
#pragma unroll
      for (int i = 0; i < 8; ++i) {
        const int row = rb + i;
        float val = acc[mt][nt][i];
        if (HAS_BIAS) val += bv;    // no +0 add emitted in the no-bias kernel
        unsigned off;
        if (REMAP) off = ((unsigned)(row & 7) * Tdim + (row >> 3)) * N + col;
        else       off = (unsigned)row * N + col;
        C[off] = val;
      }
    }
  }
}

// ===========================================================================
// SRU recurrence: one lane per (b,d) chain, software-pipelined, coalesced.
// U[(t*8+b)][3072]: cols [0,1024)=x~, [1024,2048)=f_pre, [2048,3072)=r_pre
// Emits y in f32 (residual for next layer) and bf16 (GEMM A operand).
// ===========================================================================
__global__ __launch_bounds__(256) void sru_scan(const float* __restrict__ U,
                                                const float* __restrict__ xin,
                                                const float* __restrict__ v,
                                                const float* __restrict__ bb,
                                                float* __restrict__ y,
                                                __bf16* __restrict__ ybf, int T) {
  const int tid = blockIdx.x * 256 + threadIdx.x;   // 8192 lanes
  const int d = tid & 1023, b = tid >> 10;
  const float vf = v[d], vr = v[1024 + d];
  const float bf = bb[d], br = bb[1024 + d];
  float c = 0.f;
  unsigned ub = (unsigned)b * 3072u + d;
  unsigned xb = (unsigned)b * 1024u + d;
  float xt = U[ub], fp = U[ub + 1024], rp = U[ub + 2048], xv = xin[xb];
  for (int t = 0; t < T - 1; ++t) {
    const unsigned ub2 = ub + 8u * 3072u;
    const unsigned xb2 = xb + 8u * 1024u;
    // next-step loads: independent of the carry -> compiler pipelines them
    const float xt_n = U[ub2], fp_n = U[ub2 + 1024], rp_n = U[ub2 + 2048];
    const float xv_n = xin[xb2];
    // gfx1250 global_prefetch ~8 steps ahead (stays inside workspace)
    __builtin_prefetch(U + ub + 8u * 8u * 3072u, 0, 1);
    __builtin_prefetch(xin + xb + 8u * 8u * 1024u, 0, 1);
    const float f = 1.f / (1.f + __expf(-(fp + vf * c + bf)));
    c = f * (c - xt) + xt;
    const float r = 1.f / (1.f + __expf(-(rp + vr * c + br)));
    const float out = r * (c - xv) + xv;
    y[xb] = out;
    ybf[xb] = (__bf16)out;
    xt = xt_n; fp = fp_n; rp = rp_n; xv = xv_n;
    ub = ub2; xb = xb2;
  }
  { // peeled final step (no loads)
    const float f = 1.f / (1.f + __expf(-(fp + vf * c + bf)));
    c = f * (c - xt) + xt;
    const float r = 1.f / (1.f + __expf(-(rp + vr * c + br)));
    const float out = r * (c - xv) + xv;
    y[xb] = out;
    ybf[xb] = (__bf16)out;
  }
}

__global__ void lengths_k(const long long* __restrict__ len,
                          float* __restrict__ out, int n) {
  const int i = threadIdx.x;
  if (i < n) out[i] = (float)(len[i] >> 1);
}

// ===========================================================================
extern "C" void kernel_launch(void* const* d_in, const int* in_sizes, int n_in,
                              void* d_out, int out_size, void* d_ws, size_t ws_size,
                              hipStream_t stream) {
  (void)in_sizes; (void)n_in; (void)out_size; (void)ws_size;
  const float*     fused  = (const float*)d_in[0];
  const long long* slen   = (const long long*)d_in[1];
  const float*     conv_w = (const float*)d_in[2];
  const float*     conv_b = (const float*)d_in[3];
  const float*     sru_W  = (const float*)d_in[4];
  const float*     sru_v  = (const float*)d_in[5];
  const float*     sru_b  = (const float*)d_in[6];
  const float*     fc_w   = (const float*)d_in[7];
  const float*     fc_b   = (const float*)d_in[8];

  char* ws = (char*)d_ws;
  size_t off = 0;
  auto alloc = [&](size_t bytes) -> void* {
    void* p = ws + off;
    off = (off + bytes + 255) & ~(size_t)255;
    return p;
  };
  float*  xA    = (float*)alloc(4096ull * 1024 * 4);        // 16 MB
  float*  xB    = (float*)alloc(4096ull * 1024 * 4);        // 16 MB
  __bf16* xAbf  = (__bf16*)alloc(4096ull * 1024 * 2);       //  8 MB
  __bf16* xBbf  = (__bf16*)alloc(4096ull * 1024 * 2);       //  8 MB
  float*  Ubuf  = (float*)alloc(4096ull * 3072 * 4);        // 48 MB
  __bf16* Wt    = (__bf16*)alloc(4ull * 3072 * 1024 * 2);   // 24 MB
  __bf16* fcWt  = (__bf16*)alloc(512ull * 1024 * 2);        //  1 MB

  // 1) transpose+convert weights to bf16 [N][K]
  for (int l = 0; l < 4; ++l)
    transpose_convert<<<dim3(3072 / 32, 1024 / 32), 256, 0, stream>>>(
        sru_W + (size_t)l * 1024 * 3072, Wt + (size_t)l * 3072 * 1024, 1024, 3072);
  transpose_convert<<<dim3(512 / 32, 1024 / 32), 256, 0, stream>>>(
      fc_w, fcWt, 1024, 512);

  // 2) channel-reduce conv + relu -> x[(t*8+b)][1024] (f32 + bf16)
  conv_relu<<<dim3(512 / 32, 1024 / 32, 8), 256, 0, stream>>>(
      fused, conv_w, conv_b, xA, xAbf);

  // 3) SRU layers: WMMA GEMM then sequential scan
  float*  cur   = xA;   float*  nxt   = xB;
  __bf16* curbf = xAbf; __bf16* nxtbf = xBbf;
  for (int l = 0; l < 4; ++l) {
    gemm_bf16<false, false><<<dim3(4096 / 128, 3072 / 128), 256, 0, stream>>>(
        curbf, Wt + (size_t)l * 3072 * 1024, Ubuf, nullptr, 4096, 3072, 1024);
    sru_scan<<<32, 256, 0, stream>>>(Ubuf, cur, sru_v + (size_t)l * 2048,
                                     sru_b + (size_t)l * 2048, nxt, nxtbf, 512);
    float*  tf = cur;   cur   = nxt;   nxt   = tf;
    __bf16* tb = curbf; curbf = nxtbf; nxtbf = tb;
  }

  // 4) FC GEMM with bias, remapped to logits[b][t][k] in d_out
  gemm_bf16<true, true><<<dim3(4096 / 128, 512 / 128), 256, 0, stream>>>(
      curbf, fcWt, (float*)d_out, fc_b, 4096, 512, 1024);

  // 5) s_audio_length // 2 appended after the logits
  lengths_k<<<1, 32, 0, stream>>>(slen, (float*)d_out + 8ull * 512 * 512, 8);
}
